// AdvancedDocClassificationModel_46961172414521
// MI455X (gfx1250) — matrix operand order repaired and checked by reading, not compiled, not run on hardware
//
#include <hip/hip_runtime.h>
#include <hip/hip_bf16.h>
#include <math.h>

// ---------------------------------------------------------------------------
// Model constants (from reference)
// ---------------------------------------------------------------------------
constexpr int LNUM = 4;
constexpr int Bb   = 8;
constexpr int Ss   = 512;
constexpr int Hh   = 1024;
constexpr int NHh  = 32;
constexpr int DHh  = 32;
constexpr int Ii   = 4096;
constexpr int MAXPp = 512;
constexpr int NLl  = 20;

typedef __attribute__((ext_vector_type(16))) __bf16 v16bf;
typedef __attribute__((ext_vector_type(8)))  __bf16 v8bf;
typedef __attribute__((ext_vector_type(8)))  float  v8f;

__device__ __forceinline__ v8f wmma_bf16(v16bf a, v16bf b, v8f c) {
  return __builtin_amdgcn_wmma_f32_16x16x32_bf16(
      false, a, false, b, (short)0, c, false, false);
}

__device__ __forceinline__ v16bf join8(v8bf lo, v8bf hi) {
  return __builtin_shufflevector(lo, hi, 0,1,2,3,4,5,6,7,8,9,10,11,12,13,14,15);
}

// Async global->LDS copy of 16B per lane (gfx1250 GLOBAL_LOAD_ASYNC_TO_LDS_B128,
// GVS mode: VDST = LDS byte address, VADDR = 32-bit offset, SADDR = 64-bit base).
// Tracked by ASYNCcnt.
__device__ __forceinline__ void async_b128(unsigned lds_addr, unsigned goff,
                                           const void* base) {
  asm volatile("global_load_async_to_lds_b128 %0, %1, %2"
               :: "v"(lds_addr), "v"(goff), "s"(base) : "memory");
}
__device__ __forceinline__ void wait_async0() {
  asm volatile("s_wait_asynccnt 0x0" ::: "memory");
}

// ---------------------------------------------------------------------------
// GEMM: C[M,N] = act(A[M,K] @ Wt^T + bias) with A bf16 [M,K] row-major and
// Wt bf16 [N,K] (pre-transposed weights). Block tile 128x128, 8 waves each
// computing 32x64 (2x4 WMMA tiles). Tiles staged with async-to-LDS DMA and
// double buffering. Optional fp32 and/or bf16 output.
// ---------------------------------------------------------------------------
template <int ACT>
__global__ __launch_bounds__(256) void gemm_bf16_kernel(
    const __bf16* __restrict__ A, const __bf16* __restrict__ Wt,
    const float* __restrict__ bias, float* __restrict__ Cf,
    __bf16* __restrict__ Cbf, int M, int N, int Kd)
{
  constexpr int LSTR = 40;                    // bf16 stride (80B, 16B aligned)
  __shared__ __bf16 sA[2][128 * LSTR];
  __shared__ __bf16 sB[2][128 * LSTR];

  const int tid  = threadIdx.x;
  const int w    = tid >> 5;
  const int lane = tid & 31;
  const int bm = blockIdx.y * 128;
  const int bn = blockIdx.x * 128;
  const int wm = (w & 3) * 32;
  const int wn = (w >> 2) * 64;

  v8f acc[2][4] = {};
  const int l15 = lane & 15;
  const int kb  = (lane < 16) ? 0 : 8;    // A-frag K base (elements)
  const int kkb = (lane < 16) ? 0 : 16;   // unused (B frag uses kb, like A)

  auto stage = [&](int buf, int kc) {
#pragma unroll
    for (int issue = 0; issue < 2; ++issue) {
      int idx = issue * 256 + tid;        // 512 transfers of 16B per tile
      int r = idx >> 2, seg = idx & 3;
      unsigned la = (unsigned)(size_t)(&sA[buf][r * LSTR]) + seg * 16;
      unsigned oa = (unsigned)(((size_t)(bm + r) * Kd + kc) * 2) + seg * 16;
      async_b128(la, oa, A);
      unsigned lb = (unsigned)(size_t)(&sB[buf][r * LSTR]) + seg * 16;
      unsigned ob = (unsigned)(((size_t)(bn + r) * Kd + kc) * 2) + seg * 16;
      async_b128(lb, ob, Wt);
    }
  };

  stage(0, 0);
  int buf = 0;
  for (int kc = 0; kc < Kd; kc += 32, buf ^= 1) {
    wait_async0();
    __syncthreads();
    if (kc + 32 < Kd) stage(buf ^ 1, kc + 32);

    v16bf af[2], bfr[4];
#pragma unroll
    for (int mi = 0; mi < 2; ++mi) {
      const __bf16* p = &sA[buf][(wm + mi * 16 + l15) * LSTR + kb];
      af[mi] = join8(*(const v8bf*)p, *(const v8bf*)(p + 16));
    }
#pragma unroll
    for (int ni = 0; ni < 4; ++ni) {
      const __bf16* p = &sB[buf][(wn + ni * 16 + l15) * LSTR + kb];
      bfr[ni] = join8(*(const v8bf*)p, *(const v8bf*)(p + 16));
    }
#pragma unroll
    for (int mi = 0; mi < 2; ++mi)
#pragma unroll
      for (int ni = 0; ni < 4; ++ni)
        acc[mi][ni] = wmma_bf16(af[mi], bfr[ni], acc[mi][ni]);
  }

  const int rh = (lane >= 16) ? 8 : 0;
#pragma unroll
  for (int mi = 0; mi < 2; ++mi)
#pragma unroll
    for (int ni = 0; ni < 4; ++ni) {
      int col = bn + wn + ni * 16 + l15;
      float bvv = bias ? bias[col] : 0.0f;
#pragma unroll
      for (int i = 0; i < 8; ++i) {
        int row = bm + wm + mi * 16 + i + rh;
        float v = acc[mi][ni][i] + bvv;
        if (ACT == 1) v = 0.5f * v * (1.0f + erff(v * 0.70710678118654752f));
        size_t o = (size_t)row * N + col;
        if (Cf)  Cf[o]  = v;
        if (Cbf) Cbf[o] = (__bf16)v;
      }
    }
}

// ---------------------------------------------------------------------------
// Fused flash-style attention; Q/K/V/ctx in bf16. grid = (B*NH, S/128).
// 8 waves x 16 queries. Keys in LDS chunks of 32 (K staged via async DMA).
// rel-position bias and additive mask pre-staged into LDS tables.
// ---------------------------------------------------------------------------
__global__ __launch_bounds__(256) void attn_kernel(
    const __bf16* __restrict__ Q, const __bf16* __restrict__ K,
    const __bf16* __restrict__ V, const float* __restrict__ amask,
    const float* __restrict__ rel, __bf16* __restrict__ CTX)
{
  const int bh = blockIdx.x;
  const int b  = bh / NHh;
  const int h  = bh % NHh;
  const int tid  = threadIdx.x;
  const int w    = tid >> 5;
  const int lane = tid & 31;
  const int qt = blockIdx.y * 128 + w * 16;

  __shared__ __bf16 ldsK[32 * 40];        // [key][dh]
  __shared__ __bf16 ldsV[32 * 40];        // [dh][key]
  __shared__ __bf16 ldsP[8 * 16 * 40];    // per-wave P transpose pad
  __shared__ float  ldsRel[2 * MAXPp];    // rel[(q-k)+511] for this head
  __shared__ float  ldsMask[Ss];          // (1-mask)*-1e9 per key

  const int l15 = lane & 15;
  const int kb  = (lane < 16) ? 0 : 8;
  const int kkb = (lane < 16) ? 0 : 16;
  const int rhalf = (lane >= 16) ? 8 : 0;

  // One-time LDS tables: rel diagonal for this head, additive mask
  for (int idx = tid; idx < 2 * MAXPp - 1; idx += 256)
    ldsRel[idx] = rel[(size_t)idx * DHh + h];
  for (int idx = tid; idx < Ss; idx += 256)
    ldsMask[idx] = (1.0f - amask[b * Ss + idx]) * -1e9f;

  // Q fragment (A-layout, 16x32) straight from bf16 global
  v16bf aq;
  {
    const __bf16* qp = Q + ((size_t)(b * Ss + qt + l15)) * Hh + h * DHh;
    aq = join8(*(const v8bf*)(qp + kb), *(const v8bf*)(qp + kb + 16));
  }

  v8f o0 = {}, o1 = {};
  float mr[8], ls[8];
#pragma unroll
  for (int i = 0; i < 8; ++i) { mr[i] = -3.0e38f; ls[i] = 0.0f; }

  const float scale = 0.17677669529663687f;

  for (int kc = 0; kc < Ss; kc += 32) {
    __syncthreads();
    // K chunk: async DMA, 32 rows x 64B (contiguous per row)
    if (tid < 128) {
      int key = tid >> 2, seg = tid & 3;
      unsigned la = (unsigned)(size_t)(&ldsK[key * 40]) + seg * 16;
      unsigned go = (unsigned)((((size_t)(b * Ss + kc + key)) * Hh + h * DHh) * 2) + seg * 16;
      async_b128(la, go, K);
    }
    // V chunk transposed (manual; bf16, no conversion)
    for (int idx = tid; idx < 32 * 32; idx += 256) {
      int key = idx >> 5, d = idx & 31;
      ldsV[d * 40 + key] = V[((size_t)(b * Ss + kc + key)) * Hh + h * DHh + d];
    }
    wait_async0();
    __syncthreads();

    // scores: Q(16x32) @ Kt(32x32)
    v16bf bk0, bk1;
    {
      const __bf16* p0 = &ldsK[l15 * 40 + kb];
      const __bf16* p1 = &ldsK[(16 + l15) * 40 + kb];
      bk0 = join8(*(const v8bf*)p0, *(const v8bf*)(p0 + 16));
      bk1 = join8(*(const v8bf*)p1, *(const v8bf*)(p1 + 16));
    }
    v8f c0 = {}, c1 = {};
    c0 = wmma_bf16(aq, bk0, c0);
    c1 = wmma_bf16(aq, bk1, c1);

    const int kcol0 = kc + l15;
    const float ma0 = ldsMask[kcol0];
    const float ma1 = ldsMask[kcol0 + 16];
    // rel index base: (q - k + MAXP-1); consecutive in i -> imm offsets
    const int rbase0 = qt + rhalf - kcol0 + MAXPp - 1;
    float s0[8], s1[8];
#pragma unroll
    for (int i = 0; i < 8; ++i) {
      s0[i] = (c0[i] + ldsRel[rbase0 + i])      * scale + ma0;
      s1[i] = (c1[i] + ldsRel[rbase0 - 16 + i]) * scale + ma1;
    }

#pragma unroll
    for (int i = 0; i < 8; ++i) {
      float mx = fmaxf(s0[i], s1[i]);
      for (int dd = 1; dd < 16; dd <<= 1) mx = fmaxf(mx, __shfl_xor(mx, dd, 32));
      float mnew  = fmaxf(mr[i], mx);
      float alpha = __expf(mr[i] - mnew);
      mr[i] = mnew;
      float p0 = __expf(s0[i] - mnew);
      float p1 = __expf(s1[i] - mnew);
      float rs = p0 + p1;
      for (int dd = 1; dd < 16; dd <<= 1) rs += __shfl_xor(rs, dd, 32);
      ls[i] = ls[i] * alpha + rs;
      o0[i] *= alpha;
      o1[i] *= alpha;
      int rl = i + rhalf;
      ldsP[(w * 16 + rl) * 40 + l15]      = (__bf16)p0;
      ldsP[(w * 16 + rl) * 40 + 16 + l15] = (__bf16)p1;
    }

    // ctx += P(16x32) @ V(32x32)   (wave-private pad; DS ops in order)
    v16bf ap;
    {
      const __bf16* p = &ldsP[(w * 16 + l15) * 40 + kb];
      ap = join8(*(const v8bf*)p, *(const v8bf*)(p + 16));
    }
    v16bf bv0, bv1;
    {
      const __bf16* p0 = &ldsV[l15 * 40 + kb];
      const __bf16* p1 = &ldsV[(16 + l15) * 40 + kb];
      bv0 = join8(*(const v8bf*)p0, *(const v8bf*)(p0 + 16));
      bv1 = join8(*(const v8bf*)p1, *(const v8bf*)(p1 + 16));
    }
    o0 = wmma_bf16(ap, bv0, o0);
    o1 = wmma_bf16(ap, bv1, o1);
  }

#pragma unroll
  for (int i = 0; i < 8; ++i) {
    int qrow = qt + i + rhalf;
    float inv = 1.0f / ls[i];
    __bf16* cp = CTX + ((size_t)(b * Ss + qrow)) * Hh + h * DHh;
    cp[l15]      = (__bf16)(o0[i] * inv);
    cp[16 + l15] = (__bf16)(o1[i] * inv);
  }
}

// ---------------------------------------------------------------------------
// LayerNorm + residual: x = LN(t + x); writes fp32 x and bf16 copy.
// ---------------------------------------------------------------------------
__global__ __launch_bounds__(256) void ln_res_kernel(
    const float* __restrict__ t, const float* __restrict__ g,
    const float* __restrict__ bta, float* __restrict__ x,
    __bf16* __restrict__ xbf)
{
  const int row = blockIdx.x;
  const float* tp = t + (size_t)row * Hh;
  float* xp = x + (size_t)row * Hh;
  __shared__ float red[8];

  float v[4];
  float s = 0.0f;
#pragma unroll
  for (int j = 0; j < 4; ++j) {
    int c = threadIdx.x + j * 256;
    v[j] = tp[c] + xp[c];
    s += v[j];
  }
  for (int dd = 1; dd < 32; dd <<= 1) s += __shfl_xor(s, dd, 32);
  if ((threadIdx.x & 31) == 0) red[threadIdx.x >> 5] = s;
  __syncthreads();
  float tot = 0.0f;
  for (int i = 0; i < 8; ++i) tot += red[i];
  float mu = tot / (float)Hh;

  float q = 0.0f;
#pragma unroll
  for (int j = 0; j < 4; ++j) { float dv = v[j] - mu; q += dv * dv; }
  for (int dd = 1; dd < 32; dd <<= 1) q += __shfl_xor(q, dd, 32);
  __syncthreads();
  if ((threadIdx.x & 31) == 0) red[threadIdx.x >> 5] = q;
  __syncthreads();
  float qt = 0.0f;
  for (int i = 0; i < 8; ++i) qt += red[i];
  float rinv = rsqrtf(qt / (float)Hh + 1e-12f);

#pragma unroll
  for (int j = 0; j < 4; ++j) {
    int c = threadIdx.x + j * 256;
    float y = (v[j] - mu) * rinv * g[c] + bta[c];
    xp[c] = y;
    xbf[(size_t)row * Hh + c] = (__bf16)y;
  }
}

// ---------------------------------------------------------------------------
// Weight prep: Wt[n*K + k] = (bf16)W[k*N + n]  (transpose + convert, once).
// 2-D grid: blockIdx.y = n, threads over k  (no runtime integer division).
// ---------------------------------------------------------------------------
__global__ void wprep_kernel(const float* __restrict__ W,
                             __bf16* __restrict__ Wt, int K, int N)
{
  const int n = blockIdx.y;
  const int k = blockIdx.x * 256 + threadIdx.x;
  if (k < K) Wt[(size_t)n * K + k] = (__bf16)W[(size_t)k * N + n];
}

__global__ void copy_cvt_kernel(const float* __restrict__ src,
                                float* __restrict__ dstF,
                                __bf16* __restrict__ dstB, int n)
{
  int i = blockIdx.x * blockDim.x + threadIdx.x;
  if (i < n) { float v = src[i]; dstF[i] = v; dstB[i] = (__bf16)v; }
}

// ---------------------------------------------------------------------------
// Pooling + head (tiny, bandwidth-trivial)
// ---------------------------------------------------------------------------
__global__ void pool_kernel(const float* __restrict__ X,
                            const float* __restrict__ amask,
                            float* __restrict__ pooled)
{
  const int b = blockIdx.x;
  float msum = 0.0f;
  for (int s = 0; s < Ss; ++s) msum += amask[b * Ss + s];
  for (int h = threadIdx.x; h < Hh; h += 256) {
    float sm = 0.0f, mx = -3.0e38f;
    for (int s = 0; s < Ss; ++s) {
      float v = X[((size_t)(b * Ss + s)) * Hh + h];
      float m = amask[b * Ss + s];
      sm += v * m;
      mx = fmaxf(mx, v + (1.0f - m) * -1e9f);
    }
    float* pp = pooled + (size_t)b * 3 * Hh;
    pp[h]          = sm / msum;
    pp[Hh + h]     = mx;
    pp[2 * Hh + h] = X[((size_t)(b * Ss)) * Hh + h];
  }
}

__global__ __launch_bounds__(256) void head_kernel(
    const float* __restrict__ pooled, const float* __restrict__ pW,
    const float* __restrict__ pb, const float* __restrict__ pg,
    const float* __restrict__ pbeta, const float* __restrict__ cW,
    const float* __restrict__ cb, float* __restrict__ out)
{
  constexpr int H3 = 3 * Hh;
  const int b = blockIdx.x;
  __shared__ float sp[H3];
  __shared__ float sy[Hh];
  __shared__ float red[8];

  for (int i = threadIdx.x; i < H3; i += 256) sp[i] = pooled[(size_t)b * H3 + i];
  __syncthreads();
  for (int n = threadIdx.x; n < Hh; n += 256) {
    float s = pb[n];
    for (int j = 0; j < H3; ++j) s += sp[j] * pW[(size_t)j * Hh + n];
    sy[n] = s;
  }
  __syncthreads();

  float s = 0.0f;
  for (int n = threadIdx.x; n < Hh; n += 256) s += sy[n];
  for (int dd = 1; dd < 32; dd <<= 1) s += __shfl_xor(s, dd, 32);
  if ((threadIdx.x & 31) == 0) red[threadIdx.x >> 5] = s;
  __syncthreads();
  float tot = 0.0f;
  for (int i = 0; i < 8; ++i) tot += red[i];
  float mu = tot / (float)Hh;

  float q = 0.0f;
  for (int n = threadIdx.x; n < Hh; n += 256) { float dv = sy[n] - mu; q += dv * dv; }
  for (int dd = 1; dd < 32; dd <<= 1) q += __shfl_xor(q, dd, 32);
  __syncthreads();
  if ((threadIdx.x & 31) == 0) red[threadIdx.x >> 5] = q;
  __syncthreads();
  float qt = 0.0f;
  for (int i = 0; i < 8; ++i) qt += red[i];
  float rinv = rsqrtf(qt / (float)Hh + 1e-12f);

  if (threadIdx.x < NLl) {
    float acc = cb[threadIdx.x];
    for (int n = 0; n < Hh; ++n) {
      float yn = (sy[n] - mu) * rinv * pg[n] + pbeta[n];
      acc += yn * cW[(size_t)n * NLl + threadIdx.x];
    }
    out[b * NLl + threadIdx.x] = acc;
  }
}

// ---------------------------------------------------------------------------
// Launch
// ---------------------------------------------------------------------------
extern "C" void kernel_launch(void* const* d_in, const int* in_sizes, int n_in,
                              void* d_out, int out_size, void* d_ws, size_t ws_size,
                              hipStream_t stream)
{
  (void)in_sizes; (void)n_in; (void)out_size; (void)ws_size;

  const float* hidden = (const float*)d_in[0];
  const float* amask  = (const float*)d_in[1];
  const float* Wq = (const float*)d_in[2];  const float* bq = (const float*)d_in[3];
  const float* Wk = (const float*)d_in[4];  const float* bk = (const float*)d_in[5];
  const float* Wv = (const float*)d_in[6];  const float* bv = (const float*)d_in[7];
  const float* Wo = (const float*)d_in[8];  const float* bo = (const float*)d_in[9];
  const float* ln1g = (const float*)d_in[10]; const float* ln1b = (const float*)d_in[11];
  const float* W1 = (const float*)d_in[12]; const float* b1 = (const float*)d_in[13];
  const float* W2 = (const float*)d_in[14]; const float* b2 = (const float*)d_in[15];
  const float* ln2g = (const float*)d_in[16]; const float* ln2b = (const float*)d_in[17];
  const float* relE = (const float*)d_in[18];
  const float* pW = (const float*)d_in[19]; const float* pb = (const float*)d_in[20];
  const float* pg = (const float*)d_in[21]; const float* pbb = (const float*)d_in[22];
  const float* cW = (const float*)d_in[23]; const float* cb = (const float*)d_in[24];

  const size_t BSH = (size_t)Bb * Ss * Hh;          // 4,194,304
  const size_t BSI = (size_t)Bb * Ss * Ii;          // 16,777,216
  const size_t HH2 = (size_t)Hh * Hh;
  const size_t HI  = (size_t)Hh * Ii;
  const size_t WT_PER_LAYER = 4 * HH2 + 2 * HI;     // 12,582,912 bf16

  float* X  = (float*)d_ws;
  float* T  = X + BSH;
  float* PO = T + BSH;
  __bf16* Xbf = (__bf16*)(PO + (size_t)Bb * 3 * Hh);
  __bf16* Qbf = Xbf + BSH;
  __bf16* Kbf = Qbf + BSH;
  __bf16* Vbf = Kbf + BSH;
  __bf16* Cbf = Vbf + BSH;
  __bf16* INb = Cbf + BSH;
  __bf16* WT  = INb + BSI;

  const int M = Bb * Ss;                            // 4096
  dim3 blk(256);
  dim3 g_proj(Hh / 128, M / 128);
  dim3 g_ffn1(Ii / 128, M / 128);
  dim3 g_attn(Bb * NHh, Ss / 128);
  dim3 g_ln(M);

  // One-time weight transpose+convert into bf16 arena (no runtime division)
  for (int l = 0; l < LNUM; ++l) {
    __bf16* wl = WT + (size_t)l * WT_PER_LAYER;
    wprep_kernel<<<dim3(Hh / 256, Hh), blk, 0, stream>>>(Wq + l * HH2, wl + 0 * HH2, Hh, Hh);
    wprep_kernel<<<dim3(Hh / 256, Hh), blk, 0, stream>>>(Wk + l * HH2, wl + 1 * HH2, Hh, Hh);
    wprep_kernel<<<dim3(Hh / 256, Hh), blk, 0, stream>>>(Wv + l * HH2, wl + 2 * HH2, Hh, Hh);
    wprep_kernel<<<dim3(Hh / 256, Hh), blk, 0, stream>>>(Wo + l * HH2, wl + 3 * HH2, Hh, Hh);
    wprep_kernel<<<dim3(Hh / 256, Ii), blk, 0, stream>>>(W1 + l * HI,  wl + 4 * HH2, Hh, Ii);
    wprep_kernel<<<dim3(Ii / 256, Hh), blk, 0, stream>>>(W2 + l * HI,  wl + 4 * HH2 + HI, Ii, Hh);
  }

  copy_cvt_kernel<<<(int)((BSH + 255) / 256), blk, 0, stream>>>(hidden, X, Xbf, (int)BSH);

  for (int l = 0; l < LNUM; ++l) {
    __bf16* wl = WT + (size_t)l * WT_PER_LAYER;
    const __bf16* wqT = wl + 0 * HH2;
    const __bf16* wkT = wl + 1 * HH2;
    const __bf16* wvT = wl + 2 * HH2;
    const __bf16* woT = wl + 3 * HH2;
    const __bf16* w1T = wl + 4 * HH2;
    const __bf16* w2T = wl + 4 * HH2 + HI;
    const float* rel_l = relE + (size_t)l * (2 * MAXPp - 1) * DHh;

    gemm_bf16_kernel<0><<<g_proj, blk, 0, stream>>>(Xbf, wqT, bq + l * Hh, nullptr, Qbf, M, Hh, Hh);
    gemm_bf16_kernel<0><<<g_proj, blk, 0, stream>>>(Xbf, wkT, bk + l * Hh, nullptr, Kbf, M, Hh, Hh);
    gemm_bf16_kernel<0><<<g_proj, blk, 0, stream>>>(Xbf, wvT, bv + l * Hh, nullptr, Vbf, M, Hh, Hh);

    attn_kernel<<<g_attn, blk, 0, stream>>>(Qbf, Kbf, Vbf, amask, rel_l, Cbf);

    gemm_bf16_kernel<0><<<g_proj, blk, 0, stream>>>(Cbf, woT, bo + l * Hh, T, nullptr, M, Hh, Hh);
    ln_res_kernel<<<g_ln, blk, 0, stream>>>(T, ln1g + (size_t)l * Hh, ln1b + (size_t)l * Hh, X, Xbf);

    gemm_bf16_kernel<1><<<g_ffn1, blk, 0, stream>>>(Xbf, w1T, b1 + (size_t)l * Ii, nullptr, INb, M, Ii, Hh);
    gemm_bf16_kernel<0><<<g_proj, blk, 0, stream>>>(INb, w2T, b2 + (size_t)l * Hh, T, nullptr, M, Hh, Ii);
    ln_res_kernel<<<g_ln, blk, 0, stream>>>(T, ln2g + (size_t)l * Hh, ln2b + (size_t)l * Hh, X, Xbf);
  }

  pool_kernel<<<dim3(Bb), blk, 0, stream>>>(X, amask, PO);
  head_kernel<<<dim3(Bb), blk, 0, stream>>>(PO, pW, pb, pg, pbb, cW, cb, (float*)d_out);
}
